// HAN_19954418057445
// MI455X (gfx1250) — matrix operand order repaired and checked by reading, not compiled, not run on hardware
//
#include <hip/hip_runtime.h>

typedef __bf16 bf16;
typedef __attribute__((ext_vector_type(8)))  bf16  v8bf;
typedef __attribute__((ext_vector_type(16))) bf16  v16bf;
typedef __attribute__((ext_vector_type(8)))  float v8f;

#define E_DIM 300
#define H_DIM 256
#define LSEQ  8192
#define G3    768

__device__ __forceinline__ float sigmf(float x){ return 1.0f/(1.0f+__expf(-x)); }

__device__ __forceinline__ v8f wmma_bf16(v16bf a, v16bf b, v8f c){
  // D = A(16x32 bf16) * B(32x16 bf16) + C(16x16 f32)
  return __builtin_amdgcn_wmma_f32_16x16x32_bf16(false, a, false, b, (short)0, c, false, false);
}

// Async DMA: global -> LDS, 16B per lane, tracked by ASYNCcnt.
__device__ __forceinline__ void async_ld_b128(unsigned lds_addr, const void* gaddr){
  asm volatile("global_load_async_to_lds_b128 %0, %1, off"
               :: "v"(lds_addr), "v"(gaddr) : "memory");
}
__device__ __forceinline__ void wait_async0(){
  asm volatile("s_wait_asynccnt 0x0" ::: "memory");
}

// ---------------- claim GRU (1 step, h0=0 => hidden proj = bhh) -------------
__global__ void k_claim_gru(const float* __restrict__ claim,
                            const float* __restrict__ c_wih,
                            const float* __restrict__ c_bih,
                            const float* __restrict__ c_bhh,
                            const float* __restrict__ gate_c_w,
                            float* __restrict__ hc, float* __restrict__ gc){
  __shared__ float x[E_DIM];
  __shared__ float red[H_DIM];
  int t = threadIdx.x;
  for(int k=t;k<E_DIM;k+=H_DIM) x[k]=claim[k];
  __syncthreads();
  float s0=0.f,s1=0.f,s2=0.f;
  const float* w0=&c_wih[(size_t)t*E_DIM];
  const float* w1=&c_wih[(size_t)(t+H_DIM)*E_DIM];
  const float* w2=&c_wih[(size_t)(t+2*H_DIM)*E_DIM];
  for(int k=0;k<E_DIM;k++){ float xv=x[k]; s0+=w0[k]*xv; s1+=w1[k]*xv; s2+=w2[k]*xv; }
  float r=sigmf(s0+c_bih[t]+c_bhh[t]);
  float z=sigmf(s1+c_bih[t+H_DIM]+c_bhh[t+H_DIM]);
  float n=tanhf(s2+c_bih[t+2*H_DIM]+r*c_bhh[t+2*H_DIM]);
  float h=(1.0f-z)*n;
  hc[t]=h;
  red[t]=h*gate_c_w[t];
  __syncthreads();
  for(int s=H_DIM/2;s>0;s>>=1){ if(t<s) red[t]+=red[t+s]; __syncthreads(); }
  if(t==0) gc[0]=red[0];
}

// ---------------- gx = sentences @ s_wih^T + s_bih --------------------------
__global__ void k_gx(const float* __restrict__ sents,
                     const float* __restrict__ s_wih,
                     const float* __restrict__ s_bih,
                     float* __restrict__ gx){
  __shared__ float xrow[E_DIM];
  int i=blockIdx.x;
  int j=blockIdx.y*H_DIM+threadIdx.x;
  for(int k=threadIdx.x;k<E_DIM;k+=H_DIM) xrow[k]=sents[(size_t)i*E_DIM+k];
  __syncthreads();
  const float* w=&s_wih[(size_t)j*E_DIM];
  float s=0.f;
  for(int k=0;k<E_DIM;k++) s+=w[k]*xrow[k];
  gx[(size_t)i*G3+j]=s+s_bih[j];
}

// ---------------- sequential GRU scan over 8192 steps -----------------------
__global__ void __launch_bounds__(G3) k_scan(const float* __restrict__ gx,
                                             const float* __restrict__ s_whh,
                                             const float* __restrict__ s_bhh,
                                             float* __restrict__ HS){
  __shared__ float h[H_DIM];
  __shared__ float gh[G3];
  int t=threadIdx.x;
  if(t<H_DIM) h[t]=0.f;
  float b=s_bhh[t];
  const float4* w4=(const float4*)&s_whh[(size_t)t*H_DIM];
  __syncthreads();
  for(int step=0;step<LSEQ;++step){
    if(step+1<LSEQ) __builtin_prefetch(&gx[(size_t)(step+1)*G3+t], 0, 0);
    float s=b;
    const float4* h4=(const float4*)h;
    #pragma unroll 8
    for(int k=0;k<H_DIM/4;k++){
      float4 w=w4[k], hv=h4[k];
      s+=w.x*hv.x+w.y*hv.y+w.z*hv.z+w.w*hv.w;
    }
    gh[t]=s;
    __syncthreads();
    if(t<H_DIM){
      const float* g=&gx[(size_t)step*G3];
      float r=sigmf(g[t]+gh[t]);
      float z=sigmf(g[t+H_DIM]+gh[t+H_DIM]);
      float n=tanhf(g[t+2*H_DIM]+r*gh[t+2*H_DIM]);
      float hn=(1.0f-z)*n+z*h[t];
      HS[(size_t)step*H_DIM+t]=hn;
      h[t]=hn;
    }
    __syncthreads();
  }
}

// ---------------- claim gate: hs_g (bf16) -----------------------------------
__global__ void k_gate(const float* __restrict__ HS,
                       const float* __restrict__ gate_s_w,
                       const float* __restrict__ gc,
                       const float* __restrict__ hc,
                       bf16* __restrict__ hsg){
  __shared__ float red[H_DIM];
  __shared__ float gsh;
  int i=blockIdx.x, t=threadIdx.x;
  float hv=HS[(size_t)i*H_DIM+t];
  red[t]=hv*gate_s_w[t];
  __syncthreads();
  for(int s=H_DIM/2;s>0;s>>=1){ if(t<s) red[t]+=red[t+s]; __syncthreads(); }
  if(t==0) gsh=sigmf(red[0]+gc[0]);
  __syncthreads();
  float g=gsh;
  hsg[(size_t)i*H_DIM+t]=(bf16)(g*hv+(1.0f-g)*hc[t]);
}

// ---------------- convert atten_c_w -> bf16 ---------------------------------
__global__ void k_cvt(const float* __restrict__ w, bf16* __restrict__ wb){
  int idx=blockIdx.x*H_DIM+threadIdx.x;
  wb[idx]=(bf16)w[idx];
}

// ---------------- Q = hs_g @ atten_c_w^T + atten_c_b (WMMA bf16) ------------
__global__ void __launch_bounds__(256) k_qgemm(const bf16* __restrict__ hsg,
                                               const bf16* __restrict__ Wcb,
                                               const float* __restrict__ bias,
                                               bf16* __restrict__ Qb){
  int t=threadIdx.x, wave=t>>5, lane=t&31;
  int ln=lane&15, hi=lane>>4;
  int gt=blockIdx.x*8+wave;            // 8192 wave-tiles
  int m0=(gt>>4)*16, n0=(gt&15)*16;
  v8f acc=(v8f){0,0,0,0,0,0,0,0};
  int mrow=m0+ln, nrow=n0+ln;
  #pragma unroll
  for(int c=0;c<8;c++){
    // A (16x32 bf16): lane row m=ln; two 8-element runs per lane
    const bf16* ap=&hsg[(size_t)mrow*H_DIM + c*32 + hi*16];
    v8bf lo=*(const v8bf*)ap;
    v8bf hh=*(const v8bf*)(ap+8);
    v16bf A=__builtin_shufflevector(lo,hh,0,1,2,3,4,5,6,7,8,9,10,11,12,13,14,15);
    // B (32x16 bf16): B[k][n] = W[n][k] -> lane n reads W row contiguously
    v16bf B=*(const v16bf*)&Wcb[(size_t)nrow*H_DIM + c*32 + hi*16];
    acc=wmma_bf16(A,B,acc);
  }
  float bv=bias[n0+ln];
  union {v8f v; float e[8];} u; u.v=acc;
  #pragma unroll
  for(int v=0;v<8;v++)
    Qb[(size_t)(m0+v+8*hi)*H_DIM + n0+ln]=(bf16)(u.e[v]+bv);
}

// ---------------- flash attention: h_apo = softmax(Q K^T) V -----------------
// 16 query rows per block; 8 waves split the 8192 keys (split-K) with private
// running softmax, merged via LDS. K loads feed WMMA straight from global
// (L2-resident); the V chunk is staged into per-wave LDS by async DMA
// (global_load_async_to_lds_b128) overlapped with the S-tile WMMAs, then
// transposed out of LDS into B-operand layout.
__global__ void __launch_bounds__(256) k_flash(const bf16* __restrict__ Qb,
                                               const bf16* __restrict__ KV,
                                               float* __restrict__ h_apo){
  __shared__ float Osh[16*H_DIM];
  __shared__ bf16  Pl[8][16*32];
  __shared__ bf16  Vt[8][32*H_DIM];    // per-wave 32x256 bf16 V staging (16KB each)
  __shared__ float Msh[8][16], Lsh[8][16];
  int t=threadIdx.x, wave=t>>5, lane=t&31;
  int ln=lane&15, hi=lane>>4;
  int m0=blockIdx.x*16;
  for(int k=t;k<16*H_DIM;k+=256) Osh[k]=0.0f;

  // Q rows m0..m0+15 in A-format, all 8 k-chunks, kept in registers
  v16bf Qa[8];
  {
    int mrow=m0+ln;
    #pragma unroll
    for(int c=0;c<8;c++){
      const bf16* ap=&Qb[(size_t)mrow*H_DIM + c*32 + hi*16];
      v8bf lo=*(const v8bf*)ap;
      v8bf hh=*(const v8bf*)(ap+8);
      Qa[c]=__builtin_shufflevector(lo,hh,0,1,2,3,4,5,6,7,8,9,10,11,12,13,14,15);
    }
  }
  float mcur[8], lcur[8];
  #pragma unroll
  for(int v=0;v<8;v++){ mcur[v]=-3.0e38f; lcur[v]=0.0f; }
  union F8 { v8f v; float e[8]; };
  F8 O[16];
  #pragma unroll
  for(int x=0;x<16;x++) O[x].v=(v8f){0,0,0,0,0,0,0,0};
  __syncthreads();

  unsigned vt_base=(unsigned)(unsigned long long)(void*)&Vt[wave][0];
  for(int kc=wave; kc<LSEQ/32; kc+=8){
    int key0=kc*32;
    // kick off async DMA of the V chunk (32 rows x 512B); overlaps with S WMMAs
    #pragma unroll 4
    for(int r=0;r<32;r++){
      async_ld_b128(vt_base + (unsigned)(r*H_DIM+lane*8)*2,
                    &KV[(size_t)(key0+r)*H_DIM + lane*8]);
    }
    // S[16x32] = Q @ K^T as two 16x16 C-tiles
    F8 S0,S1;
    S0.v=(v8f){0,0,0,0,0,0,0,0}; S1.v=S0.v;
    #pragma unroll
    for(int c=0;c<8;c++){
      v16bf B0=*(const v16bf*)&KV[(size_t)(key0+ln   )*H_DIM + c*32 + hi*16];
      v16bf B1=*(const v16bf*)&KV[(size_t)(key0+16+ln)*H_DIM + c*32 + hi*16];
      S0.v=wmma_bf16(Qa[c],B0,S0.v);
      S1.v=wmma_bf16(Qa[c],B1,S1.v);
    }
    // streaming softmax update (rows live in C-layout: row = v + 8*hi)
    float p0[8],p1[8];
    #pragma unroll
    for(int v=0;v<8;v++){
      float mx=fmaxf(S0.e[v],S1.e[v]);
      #pragma unroll
      for(int off=1;off<16;off<<=1) mx=fmaxf(mx,__shfl_xor(mx,off,32));
      float mnew=fmaxf(mcur[v],mx);
      float corr=__expf(mcur[v]-mnew);
      float a0=__expf(S0.e[v]-mnew);
      float a1=__expf(S1.e[v]-mnew);
      float rs=a0+a1;
      #pragma unroll
      for(int off=1;off<16;off<<=1) rs+=__shfl_xor(rs,off,32);
      lcur[v]=lcur[v]*corr+rs;
      mcur[v]=mnew;
      p0[v]=a0; p1[v]=a1;
      #pragma unroll
      for(int x=0;x<16;x++) O[x].e[v]*=corr;
    }
    // P: C-layout -> A-layout round trip through per-wave LDS tile [m][key]
    bf16* P=&Pl[wave][0];
    #pragma unroll
    for(int v=0;v<8;v++){
      int m=v+8*hi;
      P[m*32+ln]   =(bf16)p0[v];
      P[m*32+16+ln]=(bf16)p1[v];
    }
    v16bf Pa;
    {
      const bf16* pp=&P[ln*32+hi*16];
      v8bf lo=*(const v8bf*)pp;
      v8bf hh=*(const v8bf*)(pp+8);
      Pa=__builtin_shufflevector(lo,hh,0,1,2,3,4,5,6,7,8,9,10,11,12,13,14,15);
    }
    // V chunk DMA must be complete before the transpose gather
    wait_async0();
    // O[16x256] += P[16x32] @ V[32x256]; B gathered column-wise from LDS
    const bf16* vwave=&Vt[wave][0];
    #pragma unroll
    for(int x=0;x<16;x++){
      union {v16bf v; bf16 e[16];} Bv;
      const bf16* vb=&vwave[(hi*16)*H_DIM + x*16 + ln];
      #pragma unroll
      for(int e2=0;e2<16;e2++) Bv.e[e2]=vb[e2*H_DIM];
      O[x].v=wmma_bf16(Pa,Bv.v,O[x].v);
    }
  }
  // merge the 8 per-wave partial softmaxes
  if(ln==0){
    #pragma unroll
    for(int v=0;v<8;v++){ Msh[wave][v+8*hi]=mcur[v]; Lsh[wave][v+8*hi]=lcur[v]; }
  }
  __syncthreads();
  float scl[8];
  #pragma unroll
  for(int v=0;v<8;v++){
    int r=v+8*hi;
    float M=-3.0e38f;
    #pragma unroll
    for(int w=0;w<8;w++) M=fmaxf(M,Msh[w][r]);
    float L=0.0f;
    #pragma unroll
    for(int w=0;w<8;w++) L+=Lsh[w][r]*__expf(Msh[w][r]-M);
    scl[v]=__expf(mcur[v]-M)/L;
  }
  #pragma unroll
  for(int x=0;x<16;x++){
    #pragma unroll
    for(int v=0;v<8;v++)
      atomicAdd(&Osh[(v+8*hi)*H_DIM + x*16 + ln], O[x].e[v]*scl[v]);
  }
  __syncthreads();
  for(int k=t;k<16*H_DIM;k+=256)
    h_apo[(size_t)(m0+(k>>8))*H_DIM + (k&255)]=Osh[k];
}

// ---------------- h_til = tanh([HS, h_apo] @ ext_w^T + ext_b) ---------------
__global__ void k_ext(const float* __restrict__ HS,
                      const float* __restrict__ h_apo,
                      const float* __restrict__ ext_w,
                      const float* __restrict__ ext_b,
                      float* __restrict__ h_til){
  __shared__ float xrow[2*H_DIM];
  int i=blockIdx.x, j=threadIdx.x;
  xrow[j]=HS[(size_t)i*H_DIM+j];
  xrow[H_DIM+j]=h_apo[(size_t)i*H_DIM+j];
  __syncthreads();
  const float* w=&ext_w[(size_t)j*2*H_DIM];
  float s=ext_b[j];
  for(int k=0;k<2*H_DIM;k++) s+=w[k]*xrow[k];
  h_til[(size_t)i*H_DIM+j]=tanhf(s);
}

// ---------------- joint MLP: h_c_s = tanh(feat @ joint_w^T) -----------------
__global__ void k_joint(const float* __restrict__ hc,
                        const float* __restrict__ h_til,
                        const float* __restrict__ joint_w,
                        float* __restrict__ h_c_s){
  __shared__ float f[4*H_DIM];
  int i=blockIdx.x, j=threadIdx.x;
  float ht=h_til[(size_t)i*H_DIM+j], h0=hc[j];
  f[j]=h0; f[H_DIM+j]=ht; f[2*H_DIM+j]=h0*ht; f[3*H_DIM+j]=fabsf(h0-ht);
  __syncthreads();
  const float* w=&joint_w[(size_t)j*4*H_DIM];
  float s=0.f;
  for(int k=0;k<4*H_DIM;k++) s+=w[k]*f[k];
  h_c_s[(size_t)i*H_DIM+j]=tanhf(s);
}

// ---------------- entailment scores -----------------------------------------
__global__ void k_ent(const float* __restrict__ h_c_s,
                      const float* __restrict__ ent_w,
                      const float* __restrict__ ent_b,
                      float* __restrict__ e){
  int i=blockIdx.x*H_DIM+threadIdx.x;
  const float* r=&h_c_s[(size_t)i*H_DIM];
  float s=ent_b[0];
  for(int k=0;k<H_DIM;k++) s+=r[k]*ent_w[k];
  e[i]=tanhf(s);
}

// ---------------- softmax pool + final classifier ---------------------------
__global__ void k_final(const float* __restrict__ e,
                        const float* __restrict__ h_c_s,
                        const float* __restrict__ final_w,
                        const float* __restrict__ final_b,
                        float* __restrict__ out){
  __shared__ float red[H_DIM];
  __shared__ float Msh, Zsh;
  __shared__ float logits[3];
  int t=threadIdx.x;
  float mx=-3.0e38f;
  for(int i=t;i<LSEQ;i+=H_DIM) mx=fmaxf(mx,e[i]);
  red[t]=mx; __syncthreads();
  for(int s=H_DIM/2;s>0;s>>=1){ if(t<s) red[t]=fmaxf(red[t],red[t+s]); __syncthreads(); }
  if(t==0) Msh=red[0];
  __syncthreads();
  float M=Msh;
  float z=0.f;
  for(int i=t;i<LSEQ;i+=H_DIM) z+=__expf(e[i]-M);
  red[t]=z; __syncthreads();
  for(int s=H_DIM/2;s>0;s>>=1){ if(t<s) red[t]+=red[t+s]; __syncthreads(); }
  if(t==0) Zsh=red[0];
  __syncthreads();
  float Z=Zsh;
  float acc=0.f;
  for(int i=0;i<LSEQ;i++) acc+=__expf(e[i]-M)*h_c_s[(size_t)i*H_DIM+t];
  acc/=Z;
  red[t]=acc;                // h_c_S
  __syncthreads();
  if(t<3){
    float s=final_b[t];
    for(int k=0;k<H_DIM;k++) s+=final_w[t*H_DIM+k]*red[k];
    logits[t]=s;
  }
  __syncthreads();
  if(t==0){
    float m=fmaxf(logits[0],fmaxf(logits[1],logits[2]));
    float e0=__expf(logits[0]-m),e1=__expf(logits[1]-m),e2=__expf(logits[2]-m);
    float zz=e0+e1+e2;
    out[0]=e0/zz; out[1]=e1/zz; out[2]=e2/zz;
  }
}

extern "C" void kernel_launch(void* const* d_in, const int* in_sizes, int n_in,
                              void* d_out, int out_size, void* d_ws, size_t ws_size,
                              hipStream_t stream){
  (void)in_sizes; (void)n_in; (void)out_size; (void)ws_size;
  const float* claim     =(const float*)d_in[0];
  const float* sentences =(const float*)d_in[1];
  const float* c_wih     =(const float*)d_in[2];
  /* d_in[3] c_whh unused: h0 = 0 => hidden projection reduces to c_bhh   */
  const float* c_bih     =(const float*)d_in[4];
  const float* c_bhh     =(const float*)d_in[5];
  const float* s_wih     =(const float*)d_in[6];
  const float* s_whh     =(const float*)d_in[7];
  const float* s_bih     =(const float*)d_in[8];
  const float* s_bhh     =(const float*)d_in[9];
  const float* gate_s_w  =(const float*)d_in[10];
  const float* gate_c_w  =(const float*)d_in[11];
  const float* atten_c_w =(const float*)d_in[12];
  const float* atten_c_b =(const float*)d_in[13];
  /* d_in[14]/d_in[15] atten_s_w/b: add a per-row constant to the score
     matrix; row-constants cancel in softmax(axis=1) -> unused.            */
  const float* ext_w     =(const float*)d_in[16];
  const float* ext_b     =(const float*)d_in[17];
  const float* joint_w   =(const float*)d_in[18];
  const float* ent_w     =(const float*)d_in[19];
  const float* ent_b     =(const float*)d_in[20];
  const float* final_w   =(const float*)d_in[21];
  const float* final_b   =(const float*)d_in[22];
  float* out=(float*)d_out;

  char* ws=(char*)d_ws;
  size_t off=0;
  auto take=[&](size_t bytes)->char*{
    char* p=ws+off; off=(off+bytes+255)&~(size_t)255; return p;
  };
  float* hc   =(float*)take(H_DIM*sizeof(float));
  float* gc   =(float*)take(sizeof(float));
  float* gx   =(float*)take((size_t)LSEQ*G3*sizeof(float));
  float* HSp  =(float*)take((size_t)LSEQ*H_DIM*sizeof(float));
  float* h_apo=(float*)take((size_t)LSEQ*H_DIM*sizeof(float));
  float* h_til=(float*)take((size_t)LSEQ*H_DIM*sizeof(float));
  float* h_c_s=(float*)take((size_t)LSEQ*H_DIM*sizeof(float));
  float* ev   =(float*)take((size_t)LSEQ*sizeof(float));
  bf16*  hsg  =(bf16*)take((size_t)LSEQ*H_DIM*sizeof(bf16));
  bf16*  Qb   =(bf16*)take((size_t)LSEQ*H_DIM*sizeof(bf16));
  bf16*  Wcb  =(bf16*)take((size_t)H_DIM*H_DIM*sizeof(bf16));

  k_cvt      <<<dim3(H_DIM),      dim3(H_DIM), 0, stream>>>(atten_c_w, Wcb);
  k_claim_gru<<<1,                dim3(H_DIM), 0, stream>>>(claim,c_wih,c_bih,c_bhh,gate_c_w,hc,gc);
  k_gx       <<<dim3(LSEQ,3),     dim3(H_DIM), 0, stream>>>(sentences,s_wih,s_bih,gx);
  k_scan     <<<1,                dim3(G3),    0, stream>>>(gx,s_whh,s_bhh,HSp);
  k_gate     <<<dim3(LSEQ),       dim3(H_DIM), 0, stream>>>(HSp,gate_s_w,gc,hc,hsg);
  k_qgemm    <<<dim3(1024),       dim3(256),   0, stream>>>(hsg,Wcb,atten_c_b,Qb);
  k_flash    <<<dim3(LSEQ/16),    dim3(256),   0, stream>>>(Qb,hsg,h_apo);
  k_ext      <<<dim3(LSEQ),       dim3(H_DIM), 0, stream>>>(HSp,h_apo,ext_w,ext_b,h_til);
  k_joint    <<<dim3(LSEQ),       dim3(H_DIM), 0, stream>>>(hc,h_til,joint_w,h_c_s);
  k_ent      <<<dim3(LSEQ/H_DIM), dim3(H_DIM), 0, stream>>>(h_c_s,ent_w,ent_b,ev);
  k_final    <<<1,                dim3(H_DIM), 0, stream>>>(ev,h_c_s,final_w,final_b,out);
}
